// DecoderBlock_32177894981659
// MI455X (gfx1250) — compile-verified, matrix operand order checked by reading, not run on hardware
//
#include <hip/hip_runtime.h>

// ---------------------------------------------------------------------------
// CDNA5 (gfx1250) transformer decoder block: bf16 WMMA everywhere.
// B=8, S=1024, DM=512, H=8, DK=DV=64, DFF=2048
// ---------------------------------------------------------------------------

typedef __bf16 bf16;
typedef __attribute__((ext_vector_type(16))) __bf16 v16bf;
typedef __attribute__((ext_vector_type(8)))  float  v8f;

#define B_   8
#define S_   1024
#define DM_  512
#define H_   8
#define DK_  64
#define DV_  64
#define DFF_ 2048

// ---------------------------------------------------------------------------
// WMMA helper: D = A(16x32 bf16) * B(32x16 bf16) + C(16x16 f32)
// ---------------------------------------------------------------------------
__device__ __forceinline__ v8f wmma_bf16(v16bf a, v16bf b, v8f c) {
  return __builtin_amdgcn_wmma_f32_16x16x32_bf16(
      /*neg_a=*/false, a, /*neg_b=*/false, b,
      /*c_mod=*/(short)0, c, /*reuse_a=*/false, /*reuse_b=*/false);
}

// ---------------------------------------------------------------------------
// DPP16 cross-lane reductions within each 16-lane group (no LDS traffic,
// co-executes with WMMA). xor1 = quad_perm(1,0,3,2), xor2 = quad_perm(2,3,0,1),
// then ROW_HALF_MIRROR (xor7) and ROW_MIRROR (xor15): after the quad steps
// every lane holds its quad reduction, so the mirrors complete the 16-lane
// reduction. DPP16 never crosses the 16-lane row boundary.
// ---------------------------------------------------------------------------
template <int CTRL>
__device__ __forceinline__ float fdpp(float x) {
  return __int_as_float(
      __builtin_amdgcn_update_dpp(0, __float_as_int(x), CTRL, 0xF, 0xF, true));
}
__device__ __forceinline__ float row_max16(float t) {
  t = fmaxf(t, fdpp<0xB1>(t));    // quad_perm(1,0,3,2)
  t = fmaxf(t, fdpp<0x4E>(t));    // quad_perm(2,3,0,1)
  t = fmaxf(t, fdpp<0x141>(t));   // ROW_HALF_MIRROR
  t = fmaxf(t, fdpp<0x140>(t));   // ROW_MIRROR
  return t;
}
__device__ __forceinline__ float row_sum16(float t) {
  t += fdpp<0xB1>(t);
  t += fdpp<0x4E>(t);
  t += fdpp<0x141>(t);
  t += fdpp<0x140>(t);
  return t;
}

union FragU { v16bf v; float4 q[2]; };

// Fragment loader for the CDNA5 16-bit A layout (and B, with N playing M's
// role, i.e. source stored [N][K] row-major == column-major B).
// Lane L: row = L&15, hi = L>>4. Halves [0:8] hold K = hi*8+0..7,
// halves [8:16] hold K = 16+hi*8+0..7  -> two contiguous 16B loads.
__device__ __forceinline__ v16bf load_frag(const bf16* __restrict__ base,
                                           int ld, int row0, int k0) {
  const int lane = threadIdx.x & 31;
  const int m  = lane & 15;
  const int hi = lane >> 4;
  const bf16* p = base + (size_t)(row0 + m) * ld + (k0 + hi * 8);
  FragU f;
  f.q[0] = *reinterpret_cast<const float4*>(p);
  f.q[1] = *reinterpret_cast<const float4*>(p + 16);
  return f.v;
}

// ---------------------------------------------------------------------------
// fp32 -> bf16 conversion (weights)
// ---------------------------------------------------------------------------
__global__ void cvt_kernel(const float* __restrict__ src, bf16* __restrict__ dst, int n) {
  int i = blockIdx.x * blockDim.x + threadIdx.x;
  if (i < n) dst[i] = (bf16)src[i];
}

// ---------------------------------------------------------------------------
// Global LayerNorm over (S*DM) per batch; emits normalized bf16.
// grid = B, block = 256.
// ---------------------------------------------------------------------------
__global__ void ln_kernel(const float* __restrict__ x, bf16* __restrict__ out, int npb) {
  const int b = blockIdx.x;
  const float* xb = x + (size_t)b * npb;
  bf16* ob = out + (size_t)b * npb;
  float s = 0.f, s2 = 0.f;
  for (int i = threadIdx.x; i < npb; i += blockDim.x) {
    float v = xb[i];
    s += v; s2 += v * v;
  }
  __shared__ float sh_s[256], sh_q[256];
  sh_s[threadIdx.x] = s; sh_q[threadIdx.x] = s2;
  __syncthreads();
  for (int off = 128; off > 0; off >>= 1) {
    if ((int)threadIdx.x < off) {
      sh_s[threadIdx.x] += sh_s[threadIdx.x + off];
      sh_q[threadIdx.x] += sh_q[threadIdx.x + off];
    }
    __syncthreads();
  }
  const float inv_n = 1.f / (float)npb;
  const float mean = sh_s[0] * inv_n;
  const float var  = sh_q[0] * inv_n - mean * mean;
  const float rstd = rsqrtf(var + 1e-5f);
  for (int i = threadIdx.x; i < npb; i += blockDim.x)
    ob[i] = (bf16)((xb[i] - mean) * rstd);
}

// ---------------------------------------------------------------------------
// Tiled WMMA GEMM: C[M,N] = A[M,K](bf16) * W[N,K]^T(bf16) + bias, + epilogue.
// Block tile 128x64, 8 waves in 4x2 grid, 32x32 per wave (4 accumulators).
// ---------------------------------------------------------------------------
enum : int {
  EPI_BF16_QK   = 0,  // scatter to [B,H,S,64] bf16
  EPI_BF16_VT   = 1,  // scatter to [B,H,64,S] bf16 (transposed V)
  EPI_F32_RES   = 2,  // fp32 out = residual + (acc + bias)
  EPI_BF16_RELU = 3   // bf16 out = relu(acc + bias), plain [M,N]
};

template <int MODE>
__global__ void gemm_kernel(const bf16* __restrict__ A, const bf16* __restrict__ W,
                            const float* __restrict__ bias, const float* __restrict__ res,
                            void* __restrict__ outp, int M, int N, int K) {
  const int wid  = threadIdx.x >> 5;
  const int wm   = wid >> 1;            // 0..3
  const int wn   = wid & 1;             // 0..1
  const int m0   = blockIdx.y * 128 + wm * 32;
  const int n0   = blockIdx.x * 64  + wn * 32;
  const int lane = threadIdx.x & 31;
  const int hi   = lane >> 4;
  const int nn   = lane & 15;

  v8f acc[2][2] = {};
  for (int k0 = 0; k0 < K; k0 += 32) {
    v16bf a0 = load_frag(A, K, m0,      k0);
    v16bf a1 = load_frag(A, K, m0 + 16, k0);
    v16bf b0 = load_frag(W, K, n0,      k0);
    v16bf b1 = load_frag(W, K, n0 + 16, k0);
    acc[0][0] = wmma_bf16(a0, b0, acc[0][0]);
    acc[0][1] = wmma_bf16(a0, b1, acc[0][1]);
    acc[1][0] = wmma_bf16(a1, b0, acc[1][0]);
    acc[1][1] = wmma_bf16(a1, b1, acc[1][1]);
  }

#pragma unroll
  for (int i = 0; i < 2; ++i)
#pragma unroll
    for (int j = 0; j < 2; ++j)
#pragma unroll
      for (int r = 0; r < 8; ++r) {
        const int m = m0 + i * 16 + r + (hi << 3);
        const int n = n0 + j * 16 + nn;
        const float v = acc[i][j][r] + bias[n];
        if constexpr (MODE == EPI_BF16_QK) {
          const int bb = m >> 10, ss = m & 1023, hh = n >> 6, dd = n & 63;
          ((bf16*)outp)[((((size_t)(bb * 8 + hh)) << 10) + ss) * 64 + dd] = (bf16)v;
        } else if constexpr (MODE == EPI_BF16_VT) {
          const int bb = m >> 10, ss = m & 1023, hh = n >> 6, dd = n & 63;
          ((bf16*)outp)[((((size_t)(bb * 8 + hh)) * 64 + dd) << 10) + ss] = (bf16)v;
        } else if constexpr (MODE == EPI_BF16_RELU) {
          ((bf16*)outp)[(size_t)m * N + n] = (bf16)fmaxf(v, 0.f);
        } else { // EPI_F32_RES
          const size_t idx = (size_t)m * N + n;
          ((float*)outp)[idx] = res[idx] + v;
        }
      }
}

// ---------------------------------------------------------------------------
// Flash attention. Q,K: [B,H,S,DK] bf16; Vt: [B,H,DV,S] bf16 (transposed).
// out: [B,S,H*DV] bf16. Block = 4 waves = 64 query rows; wave tile = 16 rows
// x DV=64 (4 accumulators). Keys processed in 32-wide chunks with running
// max/sum (flash) softmax; P round-trips through LDS to become an A-fragment.
// Row statistics use DPP16 reductions (pure VALU, overlaps with WMMA issue).
// ---------------------------------------------------------------------------
__global__ void attn_kernel(const bf16* __restrict__ Q, const bf16* __restrict__ K,
                            const bf16* __restrict__ Vt, bf16* __restrict__ out,
                            int causal) {
  const int bh   = blockIdx.y;          // b*H + h
  const int b    = bh >> 3;
  const int h    = bh & 7;
  const int q0b  = blockIdx.x << 6;     // 64 query rows per block
  const int wid  = threadIdx.x >> 5;
  const int lane = threadIdx.x & 31;
  const int hi   = lane >> 4;
  const int nn   = lane & 15;
  const int q0   = q0b + wid * 16;

  const bf16* Qp = Q  + (size_t)bh * S_ * DK_;
  const bf16* Kp = K  + (size_t)bh * S_ * DK_;
  const bf16* Vp = Vt + (size_t)bh * DV_ * S_;

  __shared__ __align__(16) bf16 lds_p[4][16][32];

  // Q fragments for this wave's 16 rows, DK=64 -> 2 fragments.
  v16bf qa0 = load_frag(Qp, DK_, q0, 0);
  v16bf qa1 = load_frag(Qp, DK_, q0, 32);

  v8f o0 = {}, o1 = {}, o2 = {}, o3 = {};
  float mrow[8], lrow[8];
#pragma unroll
  for (int r = 0; r < 8; ++r) { mrow[r] = -1e30f; lrow[r] = 0.f; }

  const int kend = causal ? (q0b + 64) : S_;   // uniform per block
  for (int k0 = 0; k0 < kend; k0 += 32) {
    // --- scores: two 16x16 tiles over keys [k0,k0+16) and [k0+16,k0+32) ---
    v8f s0 = {}, s1 = {};
    {
      v16bf kb = load_frag(Kp, DK_, k0, 0);       s0 = wmma_bf16(qa0, kb, s0);
      kb       = load_frag(Kp, DK_, k0, 32);      s0 = wmma_bf16(qa1, kb, s0);
      kb       = load_frag(Kp, DK_, k0 + 16, 0);  s1 = wmma_bf16(qa0, kb, s1);
      kb       = load_frag(Kp, DK_, k0 + 16, 32); s1 = wmma_bf16(qa1, kb, s1);
    }
    // --- online softmax update per row ---
#pragma unroll
    for (int r = 0; r < 8; ++r) {
      const int m = r + (hi << 3);
      const int q = q0 + m;
      float v0 = s0[r] * 0.125f;   // 1/sqrt(64)
      float v1 = s1[r] * 0.125f;
      if (causal) {
        if (k0 + nn > q)      v0 = -1e30f;
        if (k0 + 16 + nn > q) v1 = -1e30f;
      }
      const float t     = row_max16(fmaxf(v0, v1));
      const float mnew  = fmaxf(mrow[r], t);
      const float scale = __expf(mrow[r] - mnew);
      const float p0 = __expf(v0 - mnew);
      const float p1 = __expf(v1 - mnew);
      const float ps = row_sum16(p0 + p1);
      lrow[r] = lrow[r] * scale + ps;
      mrow[r] = mnew;
      o0[r] *= scale; o1[r] *= scale; o2[r] *= scale; o3[r] *= scale;
      lds_p[wid][m][nn]      = (bf16)p0;
      lds_p[wid][m][16 + nn] = (bf16)p1;
    }
    __syncthreads();
    // --- P(16x32) * V(32x64): B fragments come from transposed V layout ---
    v16bf pa = load_frag(&lds_p[wid][0][0], 32, 0, 0);
    v16bf vbf;
    vbf = load_frag(Vp, S_, 0,  k0); o0 = wmma_bf16(pa, vbf, o0);
    vbf = load_frag(Vp, S_, 16, k0); o1 = wmma_bf16(pa, vbf, o1);
    vbf = load_frag(Vp, S_, 32, k0); o2 = wmma_bf16(pa, vbf, o2);
    vbf = load_frag(Vp, S_, 48, k0); o3 = wmma_bf16(pa, vbf, o3);
    __syncthreads();
  }

  // --- normalize and store to concat layout [B,S,H*DV] as bf16 ---
#pragma unroll
  for (int r = 0; r < 8; ++r) {
    const int m = r + (hi << 3);
    const int q = q0 + m;
    const float inv = 1.f / lrow[r];
    bf16* op = out + (size_t)(b * S_ + q) * (H_ * DV_) + h * DV_;
    op[nn]      = (bf16)(o0[r] * inv);
    op[16 + nn] = (bf16)(o1[r] * inv);
    op[32 + nn] = (bf16)(o2[r] * inv);
    op[48 + nn] = (bf16)(o3[r] * inv);
  }
}

// ---------------------------------------------------------------------------
// Orchestration
// ---------------------------------------------------------------------------
extern "C" void kernel_launch(void* const* d_in, const int* in_sizes, int n_in,
                              void* d_out, int out_size, void* d_ws, size_t ws_size,
                              hipStream_t stream) {
  const float* x_enc = (const float*)d_in[0];
  const float* x_dec = (const float*)d_in[1];
  const float* mq_w = (const float*)d_in[2];  const float* mq_b = (const float*)d_in[3];
  const float* mk_w = (const float*)d_in[4];  const float* mk_b = (const float*)d_in[5];
  const float* mv_w = (const float*)d_in[6];  const float* mv_b = (const float*)d_in[7];
  const float* mo_w = (const float*)d_in[8];  const float* mo_b = (const float*)d_in[9];
  const float* cq_w = (const float*)d_in[10]; const float* cq_b = (const float*)d_in[11];
  const float* ck_w = (const float*)d_in[12]; const float* ck_b = (const float*)d_in[13];
  const float* cv_w = (const float*)d_in[14]; const float* cv_b = (const float*)d_in[15];
  const float* co_w = (const float*)d_in[16]; const float* co_b = (const float*)d_in[17];
  const float* f1_w = (const float*)d_in[18]; const float* f1_b = (const float*)d_in[19];
  const float* f2_w = (const float*)d_in[20]; const float* f2_b = (const float*)d_in[21];
  float* outp = (float*)d_out;

  char* ws = (char*)d_ws;
  size_t off = 0;
  auto take = [&](size_t bytes) -> char* {
    char* p = ws + off;
    off += (bytes + 255) & ~(size_t)255;
    return p;
  };

  const int NTOK = B_ * S_;  // 8192

  // bf16 weight copies
  bf16* w_mq = (bf16*)take(512 * 512 * 2);
  bf16* w_mk = (bf16*)take(512 * 512 * 2);
  bf16* w_mv = (bf16*)take(512 * 512 * 2);
  bf16* w_mo = (bf16*)take(512 * 512 * 2);
  bf16* w_cq = (bf16*)take(512 * 512 * 2);
  bf16* w_ck = (bf16*)take(512 * 512 * 2);
  bf16* w_cv = (bf16*)take(512 * 512 * 2);
  bf16* w_co = (bf16*)take(512 * 512 * 2);
  bf16* w_f1 = (bf16*)take((size_t)DFF_ * 512 * 2);
  bf16* w_f2 = (bf16*)take((size_t)512 * DFF_ * 2);
  // activations
  bf16*  xn   = (bf16*)take((size_t)NTOK * DM_ * 2);   // normalized input (reused)
  bf16*  qb   = (bf16*)take((size_t)NTOK * 512 * 2);   // [B,H,S,DK]
  bf16*  kb   = (bf16*)take((size_t)NTOK * 512 * 2);   // [B,H,S,DK]
  bf16*  vb   = (bf16*)take((size_t)NTOK * 512 * 2);   // [B,H,DV,S]
  bf16*  cc   = (bf16*)take((size_t)NTOK * 512 * 2);   // attention concat
  float* cur  = (float*)take((size_t)NTOK * 512 * 4);  // current (fp32)
  bf16*  n2   = (bf16*)take((size_t)NTOK * 512 * 2);   // cur_n / res_n
  float* resb = (float*)take((size_t)NTOK * 512 * 4);  // result (fp32)
  bf16*  ff1  = (bf16*)take((size_t)NTOK * DFF_ * 2);  // FFN hidden

  auto cvt = [&](const float* s, bf16* d, int n) {
    cvt_kernel<<<dim3((n + 255) / 256), dim3(256), 0, stream>>>(s, d, n);
  };
  cvt(mq_w, w_mq, 512 * 512); cvt(mk_w, w_mk, 512 * 512);
  cvt(mv_w, w_mv, 512 * 512); cvt(mo_w, w_mo, 512 * 512);
  cvt(cq_w, w_cq, 512 * 512); cvt(ck_w, w_ck, 512 * 512);
  cvt(cv_w, w_cv, 512 * 512); cvt(co_w, w_co, 512 * 512);
  cvt(f1_w, w_f1, DFF_ * 512); cvt(f2_w, w_f2, 512 * DFF_);

  const dim3 blk(256);
  const dim3 g512(512 / 64, NTOK / 128);    // N=512 GEMMs
  const dim3 g2048(DFF_ / 64, NTOK / 128);  // N=2048 GEMM
  const dim3 ablk(128);
  const dim3 agrid(S_ / 64, B_ * H_);

  // ---- masked self-attention block ----
  ln_kernel<<<dim3(B_), dim3(256), 0, stream>>>(x_dec, xn, S_ * DM_);
  gemm_kernel<EPI_BF16_QK><<<g512, blk, 0, stream>>>(xn, w_mq, mq_b, nullptr, qb, NTOK, 512, 512);
  gemm_kernel<EPI_BF16_QK><<<g512, blk, 0, stream>>>(xn, w_mk, mk_b, nullptr, kb, NTOK, 512, 512);
  gemm_kernel<EPI_BF16_VT><<<g512, blk, 0, stream>>>(xn, w_mv, mv_b, nullptr, vb, NTOK, 512, 512);
  attn_kernel<<<agrid, ablk, 0, stream>>>(qb, kb, vb, cc, /*causal=*/1);
  gemm_kernel<EPI_F32_RES><<<g512, blk, 0, stream>>>(cc, w_mo, mo_b, x_dec, cur, NTOK, 512, 512);

  // ---- cross-attention block ----
  ln_kernel<<<dim3(B_), dim3(256), 0, stream>>>(cur, n2, S_ * DM_);
  ln_kernel<<<dim3(B_), dim3(256), 0, stream>>>(x_enc, xn, S_ * DM_);
  gemm_kernel<EPI_BF16_QK><<<g512, blk, 0, stream>>>(n2, w_cq, cq_b, nullptr, qb, NTOK, 512, 512);
  gemm_kernel<EPI_BF16_QK><<<g512, blk, 0, stream>>>(xn, w_ck, ck_b, nullptr, kb, NTOK, 512, 512);
  gemm_kernel<EPI_BF16_VT><<<g512, blk, 0, stream>>>(xn, w_cv, cv_b, nullptr, vb, NTOK, 512, 512);
  attn_kernel<<<agrid, ablk, 0, stream>>>(qb, kb, vb, cc, /*causal=*/0);
  gemm_kernel<EPI_F32_RES><<<g512, blk, 0, stream>>>(cc, w_co, co_b, cur, resb, NTOK, 512, 512);

  // ---- feedforward block ----
  ln_kernel<<<dim3(B_), dim3(256), 0, stream>>>(resb, n2, S_ * DM_);
  gemm_kernel<EPI_BF16_RELU><<<g2048, blk, 0, stream>>>(n2, w_f1, f1_b, nullptr, ff1, NTOK, DFF_, 512);
  gemm_kernel<EPI_F32_RES><<<g512, blk, 0, stream>>>(ff1, w_f2, f2_b, resb, outp, NTOK, 512, DFF_);
}
